// EPA_14474039788122
// MI455X (gfx1250) — matrix operand order repaired and checked by reading, not compiled, not run on hardware
//
#include <hip/hip_runtime.h>

typedef __bf16 bf16_t;
typedef __attribute__((ext_vector_type(16))) __bf16 v16bf;
typedef __attribute__((ext_vector_type(8)))  __bf16 v8bf;
typedef __attribute__((ext_vector_type(8)))  float   v8f;
typedef __attribute__((ext_vector_type(4)))  int     v4i;

union V16 { v16bf v; v8bf h[2]; };
union U64H { unsigned long long u; bf16_t h[4]; };

// --------------------------------------------------------------------------
// Optional gfx1250 async global->LDS copy path (guarded; falls back to
// register-staged copies if the builtins are not declared by this toolchain)
// --------------------------------------------------------------------------
#if defined(__has_builtin)
#if __has_builtin(__builtin_amdgcn_global_load_async_to_lds_b128) && \
    __has_builtin(__builtin_amdgcn_s_wait_asynccnt)
#define USE_ASYNC_LDS 1
#endif
#endif

#ifdef USE_ASYNC_LDS
// builtin signature (from compiler diagnostic): (v4i as1*, v4i as3*, imm, imm)
__device__ __forceinline__ void async_cp16(const void* g, void* l) {
    __builtin_amdgcn_global_load_async_to_lds_b128(
        (__attribute__((address_space(1))) v4i*)(g),
        (__attribute__((address_space(3))) v4i*)(l), 0, 0);
}
#endif

// ---------------------------------------------------------------------------
// f32 -> bf16 conversion
// ---------------------------------------------------------------------------
__global__ void convert_f32_bf16(const float* __restrict__ in,
                                 bf16_t* __restrict__ out, int n) {
    int i = blockIdx.x * blockDim.x + threadIdx.x;
    if (i < n) out[i] = (bf16_t)in[i];
}

// ---------------------------------------------------------------------------
// NT GEMM:  out[m,n] = sum_k A[m,k] * W[n,k] + bias[n]
// Block tile 128(M) x 128(N), 256 threads = 8 waves in 2x4 grid,
// wave tile 64x32 = 4x2 wmma tiles.  K stepped by 32, LDS double-buffered.
// ---------------------------------------------------------------------------
#define GT_M 128
#define GT_N 128
#define GT_K 32

__global__ __launch_bounds__(256) void gemm_bf16_nt(
    const bf16_t* __restrict__ A,
    const bf16_t* __restrict__ W,
    const float*  __restrict__ bias,
    float*        __restrict__ outF,   // optional f32 output
    bf16_t*       __restrict__ outB,   // optional bf16 output
    int M, int Nd, int K)
{
    __shared__ bf16_t As[2][GT_M * GT_K];   // 2 x 8 KB
    __shared__ bf16_t Ws[2][GT_N * GT_K];   // 2 x 8 KB

    const int tid  = threadIdx.x;
    const int wave = tid >> 5;
    const int lane = tid & 31;
    const int wm   = wave >> 2;        // 0..1
    const int wn   = wave & 3;         // 0..3
    const int l16  = lane & 15;
    const int lh   = lane >> 4;        // 0 or 1

    const int bm = blockIdx.x * GT_M;
    const int bn = blockIdx.y * GT_N;

    // per-thread staging map: 2 chunks of 16 bytes per tile
    int rS[2], cS[2];
    #pragma unroll
    for (int i = 0; i < 2; i++) {
        int ch = i * 256 + tid;        // 0..511
        rS[i] = ch >> 2;               // 0..127
        cS[i] = (ch & 3) * 8;          // 0,8,16,24
    }

    v8f acc[4][2] = {};

    // ---- prologue: stage k-chunk 0 into buffer 0 ----
    {
        #pragma unroll
        for (int i = 0; i < 2; i++) {
            const bf16_t* sa = A + (size_t)(bm + rS[i]) * K + cS[i];
            const bf16_t* sw = W + (size_t)(bn + rS[i]) * K + cS[i];
            bf16_t* da = &As[0][rS[i] * GT_K + cS[i]];
            bf16_t* dw = &Ws[0][rS[i] * GT_K + cS[i]];
#ifdef USE_ASYNC_LDS
            async_cp16(sa, da);
            async_cp16(sw, dw);
#else
            *(v8bf*)da = *(const v8bf*)sa;
            *(v8bf*)dw = *(const v8bf*)sw;
#endif
        }
#ifdef USE_ASYNC_LDS
        __builtin_amdgcn_s_wait_asynccnt(0);
#endif
        __syncthreads();
    }

    const int nk = K / GT_K;
    int cur = 0;
    for (int kk = 0; kk < nk; kk++) {
        const bool havenext = (kk + 1) < nk;
        const int  k2 = (kk + 1) * GT_K;
        const int  nxt = cur ^ 1;

        // ---- issue loads for next tile (overlap with wmma below) ----
        v8bf ra[2], rw[2];
        if (havenext) {
            #pragma unroll
            for (int i = 0; i < 2; i++) {
                const bf16_t* sa = A + (size_t)(bm + rS[i]) * K + k2 + cS[i];
                const bf16_t* sw = W + (size_t)(bn + rS[i]) * K + k2 + cS[i];
#ifdef USE_ASYNC_LDS
                async_cp16(sa, &As[nxt][rS[i] * GT_K + cS[i]]);
                async_cp16(sw, &Ws[nxt][rS[i] * GT_K + cS[i]]);
#else
                ra[i] = *(const v8bf*)sa;
                rw[i] = *(const v8bf*)sw;
#endif
            }
        }

        // ---- fragments from current buffer + 8 wmma ----
        V16 afrag[4], bfrag[2];
        #pragma unroll
        for (int mi = 0; mi < 4; mi++) {       // A layout: lane<16 K={0-7,16-23}
            int r = wm * 64 + mi * 16 + l16;   //           lane>=16 K={8-15,24-31}
            const bf16_t* p = &As[cur][r * GT_K];
            afrag[mi].h[0] = *(const v8bf*)(p + lh * 8);
            afrag[mi].h[1] = *(const v8bf*)(p + 16 + lh * 8);
        }
        #pragma unroll
        for (int ni = 0; ni < 2; ni++) {       // B layout: lane<16 K={0-7,16-23}, lane>=16 K={8-15,24-31}
            int r = wn * 32 + ni * 16 + l16;
            const bf16_t* p = &Ws[cur][r * GT_K];
            bfrag[ni].h[0] = *(const v8bf*)(p + lh * 8);
            bfrag[ni].h[1] = *(const v8bf*)(p + 16 + lh * 8);
        }
        #pragma unroll
        for (int mi = 0; mi < 4; mi++)
            #pragma unroll
            for (int ni = 0; ni < 2; ni++)
                acc[mi][ni] = __builtin_amdgcn_wmma_f32_16x16x32_bf16(
                    false, afrag[mi].v, false, bfrag[ni].v,
                    (short)0, acc[mi][ni], false, false);

        // ---- commit next tile to LDS (sync path) / drain async ----
#ifndef USE_ASYNC_LDS
        if (havenext) {
            #pragma unroll
            for (int i = 0; i < 2; i++) {
                *(v8bf*)&As[nxt][rS[i] * GT_K + cS[i]] = ra[i];
                *(v8bf*)&Ws[nxt][rS[i] * GT_K + cS[i]] = rw[i];
            }
        }
#else
        if (havenext) __builtin_amdgcn_s_wait_asynccnt(0);
#endif
        __syncthreads();
        cur = nxt;
    }

    // epilogue: C layout -> lane l, vgpr j holds (m = j + 8*lh, n = l16)
    #pragma unroll
    for (int mi = 0; mi < 4; mi++) {
        #pragma unroll
        for (int ni = 0; ni < 2; ni++) {
            #pragma unroll
            for (int j = 0; j < 8; j++) {
                int m = bm + wm * 64 + mi * 16 + j + 8 * lh;
                int n = bn + wn * 32 + ni * 16 + l16;
                float v = acc[mi][ni][j] + bias[n];
                size_t idx = (size_t)m * Nd + n;
                if (outF) outF[idx] = v;
                if (outB) outB[idx] = (bf16_t)v;
            }
        }
    }
}

// ---------------------------------------------------------------------------
// Flash attention (q == k == qk tensor).  One workgroup = 64 query rows of
// one (b,h); 4 waves x 16 rows.  KV streamed in chunks of 32 with online
// softmax.  D = 96 = 3 K-chunks of 32 (scores) = 6 N-tiles of 16 (output).
// V chunk is prefetched to registers before the score wmma burst.
// ---------------------------------------------------------------------------
__global__ __launch_bounds__(128) void flash_kernel(
    const bf16_t* __restrict__ QK,   // [B,N,C] bf16
    const bf16_t* __restrict__ Vv,   // [B,N,C] bf16
    float*        __restrict__ O,    // [B,N,C] f32 (pre-Wo out1)
    int N, int Hh, int Dd)
{
    __shared__ bf16_t vT[96 * 32];        // V^T chunk: [d][key], 6 KB
    __shared__ bf16_t Pbuf[4][16 * 32];   // per-wave P staging, 4 KB

    const int tid  = threadIdx.x;
    const int wave = tid >> 5;
    const int lane = tid & 31;
    const int l16  = lane & 15;
    const int lh   = lane >> 4;

    const int h = blockIdx.y;
    const int b = blockIdx.z;
    const int C = Hh * Dd;                        // 768
    const int q0 = blockIdx.x * 64 + wave * 16;
    const size_t baseBH = (size_t)b * N * C + (size_t)h * Dd;

    const float scale = rsqrtf((float)Dd);

    // Q fragments for 3 d-chunks (A layout, contiguous in global)
    V16 qf[3];
    {
        const bf16_t* qrow = QK + baseBH + (size_t)(q0 + l16) * C;
        #pragma unroll
        for (int c = 0; c < 3; c++) {
            qf[c].h[0] = *(const v8bf*)(qrow + c * 32 + lh * 8);
            qf[c].h[1] = *(const v8bf*)(qrow + c * 32 + 16 + lh * 8);
        }
    }

    float rmax[8], rsum[8];
    #pragma unroll
    for (int j = 0; j < 8; j++) { rmax[j] = -1e30f; rsum[j] = 0.f; }
    v8f oacc[6] = {};

    for (int k0 = 0; k0 < N; k0 += 32) {
        // ---- prefetch V chunk into registers (coalesced 8B loads) ----
        U64H vq[6];
        const bf16_t* vbase = Vv + baseBH + (size_t)k0 * C;
        #pragma unroll
        for (int i = 0; i < 6; i++) {
            int quad = i * 128 + tid;           // 0..767 (x4 halves = 3072)
            int key  = quad / 24;               // 0..31
            int dq   = (quad % 24) * 4;         // 0..92
            vq[i].u = *(const unsigned long long*)(vbase + (size_t)key * C + dq);
        }

        // ---- scores: two 16-key subtiles, 3 wmma each over D=96 ----
        v8f sacc[2] = {};
        #pragma unroll
        for (int t = 0; t < 2; t++) {
            const bf16_t* krow = QK + baseBH + (size_t)(k0 + t * 16 + l16) * C;
            #pragma unroll
            for (int c = 0; c < 3; c++) {
                V16 kf;
                const bf16_t* p = krow + c * 32;
                kf.h[0] = *(const v8bf*)(p + lh * 8);
                kf.h[1] = *(const v8bf*)(p + 16 + lh * 8);
                sacc[t] = __builtin_amdgcn_wmma_f32_16x16x32_bf16(
                    false, qf[c].v, false, kf.v, (short)0, sacc[t], false, false);
            }
        }

        // ---- online softmax (row m = j + 8*lh lives in one 16-lane half) ----
        float p[2][8], tmax[8];
        #pragma unroll
        for (int j = 0; j < 8; j++) tmax[j] = -1e30f;
        #pragma unroll
        for (int t = 0; t < 2; t++)
            #pragma unroll
            for (int j = 0; j < 8; j++) {
                float s = sacc[t][j] * scale;
                p[t][j] = s;
                tmax[j] = fmaxf(tmax[j], s);
            }
        #pragma unroll
        for (int j = 0; j < 8; j++) {
            float m = tmax[j];
            #pragma unroll
            for (int off = 1; off < 16; off <<= 1)
                m = fmaxf(m, __shfl_xor(m, off, 32));
            float nm   = fmaxf(rmax[j], m);
            float corr = __expf(rmax[j] - nm);
            rmax[j] = nm;
            float psum = 0.f;
            #pragma unroll
            for (int t = 0; t < 2; t++) {
                float e = __expf(p[t][j] - nm);
                p[t][j] = e;
                psum += e;
            }
            #pragma unroll
            for (int off = 1; off < 16; off <<= 1)
                psum += __shfl_xor(psum, off, 32);
            rsum[j] = rsum[j] * corr + psum;
            #pragma unroll
            for (int n = 0; n < 6; n++) oacc[n][j] *= corr;
        }

        // ---- C-layout -> A-layout for P via per-wave LDS round trip ----
        bf16_t* pb = &Pbuf[wave][0];
        #pragma unroll
        for (int t = 0; t < 2; t++)
            #pragma unroll
            for (int j = 0; j < 8; j++) {
                int m   = j + 8 * lh;
                int col = t * 16 + l16;
                pb[m * 32 + col] = (bf16_t)p[t][j];
            }

        __syncthreads();          // all waves done reading previous vT
        // ---- scatter V chunk transposed into LDS: vT[d][key] ----
        #pragma unroll
        for (int i = 0; i < 6; i++) {
            int quad = i * 128 + tid;
            int key  = quad / 24;
            int dq   = (quad % 24) * 4;
            #pragma unroll
            for (int j = 0; j < 4; j++)
                vT[(dq + j) * 32 + key] = vq[i].h[j];
        }
        __syncthreads();          // vT ready

        V16 pf;
        pf.h[0] = *(const v8bf*)(pb + l16 * 32 + lh * 8);
        pf.h[1] = *(const v8bf*)(pb + l16 * 32 + 16 + lh * 8);

        // ---- O += P @ Vchunk : 6 output tiles of 16 dims ----
        #pragma unroll
        for (int n = 0; n < 6; n++) {
            V16 vf;
            const bf16_t* vp = &vT[(n * 16 + l16) * 32];
            vf.h[0] = *(const v8bf*)(vp + lh * 8);
            vf.h[1] = *(const v8bf*)(vp + 16 + lh * 8);
            oacc[n] = __builtin_amdgcn_wmma_f32_16x16x32_bf16(
                false, pf.v, false, vf.v, (short)0, oacc[n], false, false);
        }
    }

    // epilogue: normalize and store f32
    #pragma unroll
    for (int j = 0; j < 8; j++) {
        int m = q0 + j + 8 * lh;
        float inv = 1.f / rsum[j];
        #pragma unroll
        for (int n = 0; n < 6; n++)
            O[baseBH + (size_t)m * C + n * 16 + l16] = oacc[n][j] * inv;
    }
}

// ---------------------------------------------------------------------------
// Branch 2 (per-token projected channel attention, tiny K=8 matmuls) fused
// with the 0.5*(out1 + x_ca) combine; emits bf16 input for the Wo GEMM.
// One block = one token, 128 threads.
// ---------------------------------------------------------------------------
__global__ __launch_bounds__(128) void branch2_kernel(
    const bf16_t* __restrict__ QK,
    const bf16_t* __restrict__ Vv,
    const float*  __restrict__ out1,
    const float*  __restrict__ We, const float* __restrict__ be,
    const float*  __restrict__ Wf, const float* __restrict__ bfp,
    bf16_t*       __restrict__ Sout)
{
    __shared__ float qs[768];
    __shared__ float vs[768];
    __shared__ float kproj[8][32];
    __shared__ float vproj[8][32];
    __shared__ float attn[32][96];

    const int tid = threadIdx.x;
    const size_t base = (size_t)blockIdx.x * 768;

    #pragma unroll
    for (int i = 0; i < 6; i++) {
        int idx = i * 128 + tid;
        qs[idx] = (float)QK[base + idx];
        vs[idx] = (float)Vv[base + idx];
    }
    __syncthreads();

    // kproj/vproj : 8x32 each; 2 elements per thread
    #pragma unroll
    for (int i = 0; i < 2; i++) {
        int idx = i * 128 + tid;
        int hh = idx >> 5;
        int e  = idx & 31;
        float accK = be[e], accV = bfp[e];
        const float* we = We + e * 96;
        const float* wf = Wf + e * 96;
        const float* q  = qs + hh * 96;
        const float* v  = vs + hh * 96;
        for (int d = 0; d < 96; d++) {
            accK += q[d] * we[d];
            accV += v[d] * wf[d];
        }
        kproj[hh][e] = accK;
        vproj[hh][e] = accV;
    }
    __syncthreads();

    // attn_sa[e][d] = softmax_d( sum_h kproj[h][e] * q[h][d] )
    if (tid < 32) {
        int e = tid;
        float mx = -1e30f;
        for (int d = 0; d < 96; d++) {
            float acc = 0.f;
            #pragma unroll
            for (int hh = 0; hh < 8; hh++) acc += kproj[hh][e] * qs[hh * 96 + d];
            attn[e][d] = acc;
            mx = fmaxf(mx, acc);
        }
        float sum = 0.f;
        for (int d = 0; d < 96; d++) {
            float ev = __expf(attn[e][d] - mx);
            attn[e][d] = ev;
            sum += ev;
        }
        float inv = 1.f / sum;
        for (int d = 0; d < 96; d++) attn[e][d] *= inv;
    }
    __syncthreads();

    // x_ca[h][d] = sum_e vproj[h][e]*attn[e][d];  S = 0.5*(out1 + x_ca)
    #pragma unroll
    for (int i = 0; i < 6; i++) {
        int idx = i * 128 + tid;
        int hh = idx / 96;
        int d  = idx % 96;
        float acc = 0.f;
        #pragma unroll
        for (int e = 0; e < 32; e++) acc += vproj[hh][e] * attn[e][d];
        float sv = 0.5f * (out1[base + idx] + acc);
        Sout[base + idx] = (bf16_t)sv;
    }
}

// ---------------------------------------------------------------------------
// Host launcher
// ---------------------------------------------------------------------------
extern "C" void kernel_launch(void* const* d_in, const int* in_sizes, int n_in,
                              void* d_out, int out_size, void* d_ws, size_t ws_size,
                              hipStream_t stream) {
    const float* x   = (const float*)d_in[0];
    const float* Wqk = (const float*)d_in[1];
    const float* bqk = (const float*)d_in[2];
    const float* Wv  = (const float*)d_in[3];
    const float* bv  = (const float*)d_in[4];
    const float* Wo  = (const float*)d_in[5];
    const float* bo  = (const float*)d_in[6];
    const float* We  = (const float*)d_in[7];
    const float* be  = (const float*)d_in[8];
    const float* Wf  = (const float*)d_in[9];
    const float* bfp = (const float*)d_in[10];

    constexpr int B = 4, N = 2048, C = 768, H = 8, D = 96;
    constexpr int M = B * N;               // 8192

    char* ws = (char*)d_ws;
    size_t off = 0;
    auto alloc = [&](size_t bytes) -> void* {
        void* p = ws + off;
        off += (bytes + 255) & ~(size_t)255;
        return p;
    };
    bf16_t* x_bf   = (bf16_t*)alloc((size_t)M * C * 2);
    bf16_t* wqk_bf = (bf16_t*)alloc((size_t)C * C * 2);
    bf16_t* wv_bf  = (bf16_t*)alloc((size_t)C * C * 2);
    bf16_t* wo_bf  = (bf16_t*)alloc((size_t)C * C * 2);
    bf16_t* qk_bf  = (bf16_t*)alloc((size_t)M * C * 2);
    bf16_t* v_bf   = (bf16_t*)alloc((size_t)M * C * 2);
    float*  out1   = (float* )alloc((size_t)M * C * 4);
    bf16_t* s_bf   = (bf16_t*)alloc((size_t)M * C * 2);

    // 1. downconvert activations + big weights
    {
        int n1 = M * C, n2 = C * C;
        convert_f32_bf16<<<(n1 + 255) / 256, 256, 0, stream>>>(x,   x_bf,   n1);
        convert_f32_bf16<<<(n2 + 255) / 256, 256, 0, stream>>>(Wqk, wqk_bf, n2);
        convert_f32_bf16<<<(n2 + 255) / 256, 256, 0, stream>>>(Wv,  wv_bf,  n2);
        convert_f32_bf16<<<(n2 + 255) / 256, 256, 0, stream>>>(Wo,  wo_bf,  n2);
    }

    dim3 ggrid(M / GT_M, C / GT_N);        // 64 x 6
    // 2. qk = x @ Wqk^T + bqk   (bf16 out)
    gemm_bf16_nt<<<ggrid, 256, 0, stream>>>(x_bf, wqk_bf, bqk, nullptr, qk_bf, M, C, C);
    // 3. v = x @ Wv^T + bv      (bf16 out)
    gemm_bf16_nt<<<ggrid, 256, 0, stream>>>(x_bf, wv_bf,  bv,  nullptr, v_bf,  M, C, C);

    // 4. flash attention -> out1 (pre-Wo)
    flash_kernel<<<dim3(N / 64, H, B), 128, 0, stream>>>(qk_bf, v_bf, out1, N, H, D);

    // 5. branch 2 + combine -> s_bf = bf16(0.5*(out1 + x_ca))
    branch2_kernel<<<dim3(M), 128, 0, stream>>>(qk_bf, v_bf, out1, We, be, Wf, bfp, s_bf);

    // 6. out = s @ Wo^T + bo  (f32 out)
    gemm_bf16_nt<<<ggrid, 256, 0, stream>>>(s_bf, wo_bf, bo, (float*)d_out, nullptr, M, C, C);
}
